// FullAttention_29557964931174
// MI455X (gfx1250) — matrix operand order repaired and checked by reading, not compiled, not run on hardware
//
#include <hip/hip_runtime.h>

#define B_ 2
#define N_ 2048
#define M_ 2048
#define H_ 8
#define D_ 64

typedef _Float16 v16h __attribute__((ext_vector_type(16)));
typedef _Float16 v8h  __attribute__((ext_vector_type(8)));
typedef float    v8f  __attribute__((ext_vector_type(8)));

__device__ __forceinline__ v8f wmma16x16x32(v16h a, v16h b, v8f c) {
  // (neg_a, A, neg_b, B, c_mod, C, reuse_a, reuse_b)
  return __builtin_amdgcn_wmma_f32_16x16x32_f16(false, a, false, b, (short)0, c,
                                                false, false);
}

// Raw hardware base-2 exponential (v_exp_f32).
__device__ __forceinline__ float fast_exp2(float x) {
#if __has_builtin(__builtin_amdgcn_exp2f)
  return __builtin_amdgcn_exp2f(x);
#else
  return exp2f(x);
#endif
}

// any-zero-byte detector for an 8-byte mask word
__device__ __forceinline__ bool has_zero_byte(unsigned long long x) {
  return ((x - 0x0101010101010101ull) & ~x & 0x8080808080808080ull) != 0ull;
}

// in: (B, rows, H, D) f32  ->  out: (B, H, rows, D) f16, scaled
__global__ __launch_bounds__(256) void pack_rows(const float* __restrict__ in,
                                                 _Float16* __restrict__ out,
                                                 int rows, float scale) {
  int idx = blockIdx.x * 256 + threadIdx.x;
  int d = idx & (D_ - 1);
  int h = (idx >> 6) & (H_ - 1);
  int r = (idx / (D_ * H_)) % rows;
  int b = idx / (D_ * H_ * rows);
  out[(((size_t)(b * H_ + h) * rows) + r) * D_ + d] = (_Float16)(in[idx] * scale);
}

// in: (B, M, H, D) f32 -> out: (B, H, D, M) f16   (V transposed)
__global__ __launch_bounds__(256) void pack_v(const float* __restrict__ in,
                                              _Float16* __restrict__ out) {
  int idx = blockIdx.x * 256 + threadIdx.x;
  int d = idx & (D_ - 1);
  int h = (idx >> 6) & (H_ - 1);
  int m = (idx / (D_ * H_)) % M_;
  int b = idx / (D_ * H_ * M_);
  out[(((size_t)(b * H_ + h) * D_) + d) * M_ + m] = (_Float16)in[idx];
}

// Flash attention forward. 4 waves/block, each wave owns 16 query rows.
// S^T = K_tile x Q^T via WMMA; P^T (C layout) feeds the PV GEMM as A directly.
// Scores are kept in the exp2 domain: Q is pre-scaled by log2(e)/sqrt(D), so
// softmax uses the native base-2 hardware transcendental (v_exp_f32).
__global__ __launch_bounds__(128) void fa_fwd(const _Float16* __restrict__ Qh,
                                              const _Float16* __restrict__ Kh,
                                              const _Float16* __restrict__ Vt,
                                              const unsigned char* __restrict__ qmask,
                                              const unsigned char* __restrict__ kvmask,
                                              float* __restrict__ out) {
  const int lane = threadIdx.x & 31;
  const int wave = threadIdx.x >> 5;
  const int bh   = blockIdx.y;         // b*H + h
  const int b    = bh >> 3;
  const int h    = bh & 7;
  const int col  = lane & 15;          // q column (QK) / d column (PV)
  const int hi   = lane >> 4;          // lane half select
  const int qRow0 = blockIdx.x * 64 + wave * 16;

  const _Float16* Qbh = Qh + (size_t)bh * N_ * D_;
  const _Float16* Kbh = Kh + (size_t)bh * M_ * D_;
  const _Float16* Vbh = Vt + (size_t)bh * D_ * M_;

  // B-layout Q (loaded once): lane holds q row `col`; K(d) = c*32 + hi*16 + [0..15]
  v16h bq0, bq1;
  {
    const _Float16* qr = Qbh + (size_t)(qRow0 + col) * D_ + hi * 16;
    bq0 = *(const v16h*)(qr);
    bq1 = *(const v16h*)(qr + 32);
  }

  float m_run = -1.0e30f;   // running max, exp2 domain
  float l_run = 0.0f;       // running denominator
  v8f acc[4] = {};          // O in C layout: 4 d-tiles of 16 cols

  for (int kv0 = 0; kv0 < M_; kv0 += 64) {
    if (kv0 + 64 < M_) {
      __builtin_prefetch(Kbh + (size_t)(kv0 + 64 + col) * D_, 0, 1);
      __builtin_prefetch(Vbh + (size_t)col * M_ + kv0 + 64, 0, 1);
    }

    // ---- S^T = K_tile(16x64) x Q^T(64x16), 4 kv sub-tiles ----
    v8f s[4];
#pragma unroll
    for (int t = 0; t < 4; ++t) {
      // A-layout K: lane holds kv row; K(d) = c*32 + hi*16 + [0..15]
      const _Float16* kr = Kbh + (size_t)(kv0 + t * 16 + col) * D_;
      v8f c0 = {};
      {
        v8h lo  = *(const v8h*)(kr + hi * 16);
        v8h hi8 = *(const v8h*)(kr + 8 + hi * 16);
        v16h a = __builtin_shufflevector(lo, hi8, 0,1,2,3,4,5,6,7,8,9,10,11,12,13,14,15);
        c0 = wmma16x16x32(a, bq0, c0);
      }
      {
        v8h lo  = *(const v8h*)(kr + 32 + hi * 16);
        v8h hi8 = *(const v8h*)(kr + 40 + hi * 16);
        v16h a = __builtin_shufflevector(lo, hi8, 0,1,2,3,4,5,6,7,8,9,10,11,12,13,14,15);
        c0 = wmma16x16x32(a, bq1, c0);
      }
      s[t] = c0;
    }

    // ---- kv mask: wave-uniform fast path (selects skipped when all-true) ----
    {
      const unsigned long long* mp =
          (const unsigned long long*)(kvmask + (size_t)b * M_ + kv0 + hi * 8);
      unsigned long long mb[4];
#pragma unroll
      for (int t = 0; t < 4; ++t) mb[t] = mp[t * 2];
      bool anyz = has_zero_byte(mb[0] & mb[1] & mb[2] & mb[3]);
      if (__any((int)anyz)) {
#pragma unroll
        for (int t = 0; t < 4; ++t)
#pragma unroll
          for (int v = 0; v < 8; ++v)
            if (((mb[t] >> (8 * v)) & 0xffull) == 0) s[t][v] = -1.0e9f;
      }
    }

    // ---- online softmax (per q row = per lane pair), base-2 domain ----
    float mx = m_run;
#pragma unroll
    for (int t = 0; t < 4; ++t)
#pragma unroll
      for (int v = 0; v < 8; ++v) mx = fmaxf(mx, s[t][v]);
    mx = fmaxf(mx, __shfl_xor(mx, 16));
    float fct = fast_exp2(m_run - mx);
    float ps = 0.0f;
#pragma unroll
    for (int t = 0; t < 4; ++t)
#pragma unroll
      for (int v = 0; v < 8; ++v) {
        float e = fast_exp2(s[t][v] - mx);
        s[t][v] = e;
        ps += e;
      }
    ps += __shfl_xor(ps, 16);
    l_run = l_run * fct + ps;
    m_run = mx;

    // ---- rescale O accumulators: acc row (hi*8+v) needs factor from lane (hi*8+v) ----
#pragma unroll
    for (int v = 0; v < 8; ++v) {
      float fv = __shfl(fct, hi * 8 + v);
      acc[0][v] *= fv; acc[1][v] *= fv; acc[2][v] *= fv; acc[3][v] *= fv;
    }

    // ---- P^T (C layout) -> P as A-matrix chunks, in-lane repack only ----
    v16h pa0 = {}, pa1 = {};
#pragma unroll
    for (int j = 0; j < 8; ++j) {
      pa0[j]     = (_Float16)s[0][j];
      pa0[8 + j] = (_Float16)s[1][j];
      pa1[j]     = (_Float16)s[2][j];
      pa1[8 + j] = (_Float16)s[3][j];
    }

    // ---- O += P(16x64) x V(64x64): B from transposed Vt, table k order per lane ----
#pragma unroll
    for (int dt = 0; dt < 4; ++dt) {
      const _Float16* vr = Vbh + (size_t)(dt * 16 + col) * M_ + kv0 + hi * 8;
      v16h bv0 = __builtin_shufflevector(*(const v8h*)(vr), *(const v8h*)(vr + 16), 0,1,2,3,4,5,6,7,8,9,10,11,12,13,14,15);
      v16h bv1 = __builtin_shufflevector(*(const v8h*)(vr + 32), *(const v8h*)(vr + 48), 0,1,2,3,4,5,6,7,8,9,10,11,12,13,14,15);
      acc[dt] = wmma16x16x32(pa0, bv0, acc[dt]);
      acc[dt] = wmma16x16x32(pa1, bv1, acc[dt]);
    }
  }

  // ---- epilogue: normalize by l, apply q_mask, scatter-store f32 output ----
  float linv = (l_run > 0.0f) ? (1.0f / l_run) : 0.0f;
  unsigned long long qmb =
      *(const unsigned long long*)(qmask + (size_t)b * N_ + qRow0 + hi * 8);
  float srow[8];
  bool qz = has_zero_byte(qmb);
  if (__any((int)qz)) {
#pragma unroll
    for (int v = 0; v < 8; ++v) {
      float lv = __shfl(linv, hi * 8 + v);
      srow[v] = (((qmb >> (8 * v)) & 0xffull) != 0) ? lv : 0.0f;
    }
  } else {
#pragma unroll
    for (int v = 0; v < 8; ++v) srow[v] = __shfl(linv, hi * 8 + v);
  }
#pragma unroll
  for (int dt = 0; dt < 4; ++dt) {
#pragma unroll
    for (int v = 0; v < 8; ++v) {
      int qr = qRow0 + hi * 8 + v;
      out[(((size_t)(b * N_ + qr) * H_) + h) * D_ + dt * 16 + col] =
          acc[dt][v] * srow[v];
    }
  }
}

extern "C" void kernel_launch(void* const* d_in, const int* in_sizes, int n_in,
                              void* d_out, int out_size, void* d_ws, size_t ws_size,
                              hipStream_t stream) {
  const float* q = (const float*)d_in[0];
  const float* k = (const float*)d_in[1];
  const float* v = (const float*)d_in[2];
  const unsigned char* qmask  = (const unsigned char*)d_in[3];
  const unsigned char* kvmask = (const unsigned char*)d_in[4];
  float* out = (float*)d_out;

  const size_t elems = (size_t)B_ * N_ * H_ * D_;  // N_ == M_, same count for K/V
  _Float16* Qh = (_Float16*)d_ws;
  _Float16* Kh = Qh + elems;
  _Float16* Vt = Kh + elems;

  // Q scale folds 1/sqrt(D) AND log2(e) so the softmax runs in the exp2 domain.
  const float qscale = 0.125f * 1.44269504088896340736f;

  dim3 pgrid((unsigned)(elems / 256));
  pack_rows<<<pgrid, 256, 0, stream>>>(q, Qh, N_, qscale);
  pack_rows<<<pgrid, 256, 0, stream>>>(k, Kh, M_, 1.0f);
  pack_v  <<<pgrid, 256, 0, stream>>>(v, Vt);

  dim3 grid(N_ / 64, B_ * H_);
  fa_fwd<<<grid, 128, 0, stream>>>(Qh, Kh, Vt, qmask, kvmask, out);
}
